// WorkingMemory_24292335026471
// MI455X (gfx1250) — matrix operand order repaired and checked by reading, not compiled, not run on hardware
//
#include <hip/hip_runtime.h>
#include <hip/hip_bf16.h>

typedef __attribute__((ext_vector_type(16))) _Float16 v16h;
typedef __attribute__((ext_vector_type(8)))  _Float16 v8h;
typedef __attribute__((ext_vector_type(8)))  float    v8f;

#define BSZ 64
#define PQ  256          // span length
#define DM  512          // D == D_wm
#define WC  1024         // cache length
#define HH  8            // heads
#define HDD 64           // head dim
#define LC  (WC + PQ)    // 1280 total keys
#define ATT_SCALE 0.125f // 1/sqrt(64)

// ---- workspace layout (bytes); total ~194 MiB ----
#define WS_WT   ((size_t)0)                                   // 4 x (512*512) f16 transposed weights
#define WS_Q    ((size_t)4 * 512 * 512 * 2)                   // q f16 (16384 x 512)
#define WS_KC   (WS_Q  + (size_t)BSZ * PQ * DM * 2)           // Kc f16 (BSZ x 1280 x 512)
#define WS_VCT  (WS_KC + (size_t)BSZ * LC * DM * 2)           // VcT f16 (BSZ x H x 64 x 1280)
#define WS_AO   (WS_VCT + (size_t)BSZ * HH * HDD * LC * 2)    // attn out f16 (16384 x 512)

__device__ __forceinline__ v8f wmma_f16(v16h a, v16h b, v8f c) {
  return __builtin_amdgcn_wmma_f32_16x16x32_f16(false, a, false, b, (short)0, c, false, false);
}

// A-matrix 16x32 f16 fragment from row-major f16 src (contiguous k order, same as the B loader):
// lanes 0-15: row M=lane, K 0..15; lanes 16-31: row M=lane-16, K 16..31
__device__ __forceinline__ v16h load_a16x32_f16(const _Float16* src, int ld, int row0, int k0, int lane) {
  int m = lane & 15, s = (lane >> 4) & 1;
  const _Float16* p = src + (size_t)(row0 + m) * ld + k0 + s * 16;
  v8h lo = *(const v8h*)p;
  v8h hi = *(const v8h*)(p + 8);
  v16h r;
#pragma unroll
  for (int i = 0; i < 8; ++i) { r[i] = lo[i]; r[i + 8] = hi[i]; }
  return r;
}

// Same A layout, converting from f32 source in registers
__device__ __forceinline__ v16h load_a16x32_f32(const float* src, int ld, int row0, int k0, int lane) {
  int m = lane & 15, s = (lane >> 4) & 1;
  const float* p = src + (size_t)(row0 + m) * ld + k0 + s * 16;
  v16h r;
#pragma unroll
  for (int i = 0; i < 8; ++i) { r[i] = (_Float16)p[i]; r[i + 8] = (_Float16)p[i + 8]; }
  return r;
}

// B-matrix 32x16 f16 fragment; src is column-major-friendly [n][k] (row stride ld):
// lanes 0-15: col N=lane, K 0..15; lanes 16-31: col N=lane-16, K 16..31
__device__ __forceinline__ v16h load_b32x16_f16(const _Float16* src, int ld, int n0, int k0, int lane) {
  int n = lane & 15, s = (lane >> 4) & 1;
  const _Float16* p = src + (size_t)(n0 + n) * ld + k0 + s * 16;
  v8h a = *(const v8h*)p;
  v8h b = *(const v8h*)(p + 8);
  v16h r;
#pragma unroll
  for (int i = 0; i < 8; ++i) { r[i] = a[i]; r[i + 8] = b[i]; }
  return r;
}

// ---------------- prep: LDS-tiled transpose+convert (coalesced both sides) ----------------
__global__ void prep_wt_t(const float* __restrict__ wq, const float* __restrict__ wk,
                          const float* __restrict__ wv, const float* __restrict__ wo,
                          _Float16* __restrict__ wt) {
  // grid: 4 mats * 16 * 16 tiles = 1024 blocks of 256
  __shared__ float tile[32][33];
  int bid = blockIdx.x;
  int tc = bid & 15;          // n tile
  int tr = (bid >> 4) & 15;   // k tile
  int m  = bid >> 8;
  const float* src = (m == 0) ? wq : (m == 1) ? wk : (m == 2) ? wv : wo;  // [k][n]
  _Float16* d = wt + (size_t)m * 262144;                                  // [n][k]
  int lx = threadIdx.x & 31, ly = threadIdx.x >> 5;
#pragma unroll
  for (int i = 0; i < 4; ++i) {
    int k = tr * 32 + ly * 4 + i;
    tile[ly * 4 + i][lx] = src[(size_t)k * 512 + tc * 32 + lx];
  }
  __syncthreads();
#pragma unroll
  for (int i = 0; i < 4; ++i) {
    int n = tc * 32 + ly * 4 + i;
    d[(size_t)n * 512 + tr * 32 + lx] = (_Float16)tile[lx][ly * 4 + i];
  }
}

__global__ void prep_kc(const float* __restrict__ wmk, _Float16* __restrict__ kcc) {
  size_t tid = (size_t)blockIdx.x * 256 + threadIdx.x;  // BSZ*WC*DM
  int c = (int)(tid & 511);
  size_t rw = tid >> 9;
  int w = (int)(rw & 1023);
  int b = (int)(rw >> 10);
  kcc[((size_t)b * LC + w) * DM + c] = (_Float16)wmk[tid];
}

__global__ void prep_vct_t(const float* __restrict__ wmv, _Float16* __restrict__ vct) {
  // per (b,h): transpose (1024 w x 64 hd, row stride 512) -> (64 hd x 1280ld)
  // grid: BSZ*HH*32*2 = 32768 blocks of 256
  __shared__ float tile[32][33];
  int bid = blockIdx.x;
  int tc = bid & 1;           // hd tile (64/32)
  int tr = (bid >> 1) & 31;   // w tile (1024/32)
  int h  = (bid >> 6) & 7;
  int b  = bid >> 9;
  int lx = threadIdx.x & 31, ly = threadIdx.x >> 5;
  const float* s = wmv + ((size_t)b * WC) * DM + h * HDD;
#pragma unroll
  for (int i = 0; i < 4; ++i) {
    int w = tr * 32 + ly * 4 + i;
    tile[ly * 4 + i][lx] = s[(size_t)w * DM + tc * 32 + lx];
  }
  __syncthreads();
  _Float16* d = vct + ((size_t)(b * HH + h) * HDD) * LC;
#pragma unroll
  for (int i = 0; i < 4; ++i) {
    int hd = tc * 32 + ly * 4 + i;
    d[(size_t)hd * LC + tr * 32 + lx] = (_Float16)tile[lx][ly * 4 + i];
  }
}

// ---------------- QKV projection: Y = X @ W + b, f16 outputs ----------------
__global__ void qkv_gemm(const float* __restrict__ x,
                         const _Float16* __restrict__ wtq, const _Float16* __restrict__ wtk,
                         const _Float16* __restrict__ wtv,
                         const float* __restrict__ bq, const float* __restrict__ bk,
                         const float* __restrict__ bv,
                         _Float16* __restrict__ qf, _Float16* __restrict__ kcc,
                         _Float16* __restrict__ vct) {
  int lane = threadIdx.x & 31;
  int gw = blockIdx.x * 8 + (threadIdx.x >> 5);  // 24576 wave tasks
  int rt = gw / 24;
  int rem = gw % 24;
  int mat = rem >> 3;
  int strip = rem & 7;
  int rowbase = rt * 16;
  const _Float16* wt = (mat == 0) ? wtq : (mat == 1) ? wtk : wtv;
  const float* bias = (mat == 0) ? bq : (mat == 1) ? bk : bv;
  int nl = lane & 15, shi = lane >> 4;

  v8f acc[4] = {};
  for (int kk = 0; kk < DM; kk += 32) {
    if (kk + 32 < DM)
      __builtin_prefetch(x + (size_t)(rowbase + nl) * DM + kk + 32 + shi * 16, 0, 3);
    v16h a = load_a16x32_f32(x, DM, rowbase, kk, lane);
#pragma unroll
    for (int t = 0; t < 4; ++t) {
      v16h bf = load_b32x16_f16(wt, DM, strip * 64 + 16 * t, kk, lane);
      acc[t] = wmma_f16(a, bf, acc[t]);
    }
  }

  int bidx = rowbase >> 8;
  int pp0 = rowbase & 255;
#pragma unroll
  for (int t = 0; t < 4; ++t) {
    int col = strip * 64 + 16 * t + nl;
    float bb = bias[col];
    if (mat == 2) {
      // store V transposed: VcT[b][h=strip][hd=16t+nl][key], 8 consecutive keys -> one B128
      v8h st;
#pragma unroll
      for (int r = 0; r < 8; ++r) st[r] = (_Float16)(acc[t][r] + bb);
      size_t off = ((size_t)(bidx * HH + strip) * HDD + 16 * t + nl) * LC + WC + pp0 + 8 * shi;
      *(v8h*)(vct + off) = st;
    } else {
#pragma unroll
      for (int r = 0; r < 8; ++r) {
        int row = rowbase + r + 8 * shi;   // C layout: VGPR r holds M = r (+8 for hi lanes)
        _Float16 v = (_Float16)(acc[t][r] + bb);
        if (mat == 0) {
          qf[(size_t)row * DM + col] = v;
        } else {
          int b2 = row >> 8, pp = row & 255;
          kcc[((size_t)b2 * LC + WC + pp) * DM + col] = v;
        }
      }
    }
  }
}

// ---------------- attention helpers ----------------
__device__ __forceinline__ float mask_s(float s, int key, int pq, int rst,
                                        const unsigned char* vb, int wp) {
  bool inCache = key < WC;
  bool okc = (rst == 0) && (vb[key & (WC - 1)] != 0) && (((key - wp) & (WC - 1)) > pq);
  bool oks = (key - WC) <= pq;
  bool ok = inCache ? okc : oks;
  return ok ? s * ATT_SCALE : -1.0e30f;
}

// online-softmax update for one 16-query group; returns P^T B-fragment for the PV WMMAs
__device__ __forceinline__ v16h softmax_update(const v8f& st0, const v8f& st1, int kc, int pq,
                                               int shi, int rst, const unsigned char* vb, int wp,
                                               float& m_run, float& l_run, v8f* o) {
  float pv[16];
#pragma unroll
  for (int r = 0; r < 8; ++r) {
    int key0 = kc + r + 8 * shi;
    pv[r]     = mask_s(st0[r], key0,      pq, rst, vb, wp);
    pv[8 + r] = mask_s(st1[r], key0 + 16, pq, rst, vb, wp);
  }
  float mloc = pv[0];
#pragma unroll
  for (int i = 1; i < 16; ++i) mloc = fmaxf(mloc, pv[i]);
  mloc = fmaxf(mloc, __shfl_xor(mloc, 16, 32));
  float m_new = fmaxf(m_run, mloc);
  float fsc = __expf(m_run - m_new);
  float ssum = 0.0f;
#pragma unroll
  for (int i = 0; i < 16; ++i) { pv[i] = __expf(pv[i] - m_new); ssum += pv[i]; }
  ssum += __shfl_xor(ssum, 16, 32);
  l_run = l_run * fsc + ssum;
  m_run = m_new;
#pragma unroll
  for (int t = 0; t < 4; ++t)
#pragma unroll
    for (int r = 0; r < 8; ++r) o[t][r] *= fsc;

  v16h pf;
#pragma unroll
  for (int r = 0; r < 8; ++r) {
    float o0v = __shfl_xor(pv[r], 16, 32);      // partner's tile0 rows
    float o1v = __shfl_xor(pv[8 + r], 16, 32);  // partner's tile1 rows
    float lo = shi ? o1v : pv[r];               // keys kb+0..7
    float hi = shi ? pv[8 + r] : o0v;           // keys kb+8..15
    pf[r] = (_Float16)lo;
    pf[8 + r] = (_Float16)hi;
  }
  return pf;
}

__device__ __forceinline__ void store_o(_Float16* __restrict__ ao, int b, int pq, int h, int shi,
                                        const v8f* o, float l_run) {
  float inv = (l_run > 0.0f) ? (1.0f / l_run) : 0.0f;
  _Float16* arow = ao + ((size_t)(b * PQ + pq) * DM + h * HDD);
#pragma unroll
  for (int th = 0; th < 4; ++th) {
    v8h st;
#pragma unroll
    for (int r = 0; r < 8; ++r) st[r] = (_Float16)(o[th][r] * inv);
    *(v8h*)(arow + 16 * th + 8 * shi) = st;
  }
}

// ---------------- attention: flash-style, 32 queries/wave, S^T = K@Q^T, O^T = V^T@P^T ----------
__global__ void attn_kernel(const _Float16* __restrict__ qf, const _Float16* __restrict__ kcc,
                            const _Float16* __restrict__ vct, _Float16* __restrict__ ao,
                            const unsigned char* __restrict__ reset_mask,
                            const unsigned char* __restrict__ wm_valid,
                            const int* __restrict__ wm_ptr) {
  int lane = threadIdx.x & 31;
  int gw = blockIdx.x * 8 + (threadIdx.x >> 5);  // 4096 wave tasks: b*64 + h*8 + qt
  int b = gw >> 6;
  int h = (gw >> 3) & 7;
  int p0 = (gw & 7) << 5;                        // 32-query tile
  int nl = lane & 15, shi = lane >> 4;

  int rst = reset_mask[(size_t)b * PQ] ? 1 : 0;
  int wp = wm_ptr[b];
  const unsigned char* vb = wm_valid + (size_t)b * WC;

  const _Float16* qB = qf + (size_t)b * PQ * DM;
  const _Float16* kB = kcc + (size_t)b * LC * DM;
  const _Float16* vT = vct + ((size_t)(b * HH + h) * HDD) * LC;

  // Q^T B-fragments for two 16-query groups x two hd halves
  v16h qb00 = load_b32x16_f16(qB, DM, p0,      h * HDD,      lane);
  v16h qb01 = load_b32x16_f16(qB, DM, p0,      h * HDD + 32, lane);
  v16h qb10 = load_b32x16_f16(qB, DM, p0 + 16, h * HDD,      lane);
  v16h qb11 = load_b32x16_f16(qB, DM, p0 + 16, h * HDD + 32, lane);

  float m_run0 = -__builtin_inff(), l_run0 = 0.0f;
  float m_run1 = -__builtin_inff(), l_run1 = 0.0f;
  v8f o0[4] = {}, o1[4] = {};

  for (int kc = 0; kc < LC; kc += 32) {
    if (kc + 32 < LC) {
      __builtin_prefetch(kB + (size_t)(kc + 32 + lane) * DM + h * HDD, 0, 3);
      __builtin_prefetch(vT + (size_t)(lane >> 3) * 16 * LC + ((size_t)(lane & 7) * 2) * LC + kc + 32, 0, 3);
    }
    // K A-fragments: 2 key tiles x 2 hd halves, each reused by both query groups
    v16h ka00 = load_a16x32_f16(kB, DM, kc,      h * HDD,      lane);
    v16h ka01 = load_a16x32_f16(kB, DM, kc,      h * HDD + 32, lane);
    v16h ka10 = load_a16x32_f16(kB, DM, kc + 16, h * HDD,      lane);
    v16h ka11 = load_a16x32_f16(kB, DM, kc + 16, h * HDD + 32, lane);

    v8f s00 = {}, s10 = {}, s01 = {}, s11 = {};  // s[keytile][qgroup]
    s00 = wmma_f16(ka00, qb00, s00); s00 = wmma_f16(ka01, qb01, s00);
    s01 = wmma_f16(ka00, qb10, s01); s01 = wmma_f16(ka01, qb11, s01);
    s10 = wmma_f16(ka10, qb00, s10); s10 = wmma_f16(ka11, qb01, s10);
    s11 = wmma_f16(ka10, qb10, s11); s11 = wmma_f16(ka11, qb11, s11);

    v16h pf0 = softmax_update(s00, s10, kc, p0 + nl,      shi, rst, vb, wp, m_run0, l_run0, o0);
    v16h pf1 = softmax_update(s01, s11, kc, p0 + 16 + nl, shi, rst, vb, wp, m_run1, l_run1, o1);

    // O^T += V^T @ P^T ; each V fragment reused by both query groups
#pragma unroll
    for (int th = 0; th < 4; ++th) {
      v16h va = load_a16x32_f16(vT, LC, 16 * th, kc, lane);
      o0[th] = wmma_f16(va, pf0, o0[th]);
      o1[th] = wmma_f16(va, pf1, o1[th]);
    }
  }

  store_o(ao, b, p0 + nl,      h, shi, o0, l_run0);
  store_o(ao, b, p0 + 16 + nl, h, shi, o1, l_run1);
}

// ---------------- output projection: y = AO @ Wo + bo (f32 out) ----------------
__global__ void out_gemm(const _Float16* __restrict__ ao, const _Float16* __restrict__ wot,
                         const float* __restrict__ bo, float* __restrict__ y) {
  int lane = threadIdx.x & 31;
  int gw = blockIdx.x * 8 + (threadIdx.x >> 5);  // 8192 wave tasks
  int rt = gw >> 3, strip = gw & 7;
  int rowbase = rt * 16;
  int nl = lane & 15, shi = lane >> 4;

  v8f acc[4] = {};
  for (int kk = 0; kk < DM; kk += 32) {
    if (kk + 32 < DM)
      __builtin_prefetch(ao + (size_t)(rowbase + nl) * DM + kk + 32 + shi * 16, 0, 3);
    v16h a = load_a16x32_f16(ao, DM, rowbase, kk, lane);
#pragma unroll
    for (int t = 0; t < 4; ++t) {
      v16h bf = load_b32x16_f16(wot, DM, strip * 64 + 16 * t, kk, lane);
      acc[t] = wmma_f16(a, bf, acc[t]);
    }
  }
#pragma unroll
  for (int t = 0; t < 4; ++t) {
    int col = strip * 64 + 16 * t + nl;
    float bb = bo[col];
#pragma unroll
    for (int r = 0; r < 8; ++r) {
      int row = rowbase + r + 8 * shi;
      y[(size_t)row * 512 + col] = acc[t][r] + bb;
    }
  }
}

extern "C" void kernel_launch(void* const* d_in, const int* in_sizes, int n_in,
                              void* d_out, int out_size, void* d_ws, size_t ws_size,
                              hipStream_t stream) {
  (void)in_sizes; (void)n_in; (void)out_size; (void)ws_size;
  const float* x            = (const float*)d_in[0];
  const unsigned char* rst  = (const unsigned char*)d_in[1];
  const float* wmK          = (const float*)d_in[2];
  const float* wmV          = (const float*)d_in[3];
  const unsigned char* wval = (const unsigned char*)d_in[4];
  const int* wptr           = (const int*)d_in[5];
  const float* Wq = (const float*)d_in[6];
  const float* bq = (const float*)d_in[7];
  const float* Wk = (const float*)d_in[8];
  const float* bk = (const float*)d_in[9];
  const float* Wv = (const float*)d_in[10];
  const float* bv = (const float*)d_in[11];
  const float* Wo = (const float*)d_in[12];
  const float* bo = (const float*)d_in[13];
  float* y = (float*)d_out;

  char* ws = (char*)d_ws;
  _Float16* wt  = (_Float16*)(ws + WS_WT);   // [4][512][512] transposed f16
  _Float16* qf  = (_Float16*)(ws + WS_Q);
  _Float16* kcc = (_Float16*)(ws + WS_KC);
  _Float16* vct = (_Float16*)(ws + WS_VCT);
  _Float16* ao  = (_Float16*)(ws + WS_AO);

  prep_wt_t <<<1024,   256, 0, stream>>>(Wq, Wk, Wv, Wo, wt);
  prep_kc   <<<131072, 256, 0, stream>>>(wmK, kcc);
  prep_vct_t<<<32768,  256, 0, stream>>>(wmV, vct);
  qkv_gemm  <<<3072,   256, 0, stream>>>(x, wt, wt + 262144, wt + 2 * 262144,
                                         bq, bk, bv, qf, kcc, vct);
  attn_kernel<<<512,   256, 0, stream>>>(qf, kcc, vct, ao, rst, wval, wptr);
  out_gemm  <<<1024,   256, 0, stream>>>(ao, wt + 3 * 262144, bo, y);
}
